// Attention_74852690035363
// MI455X (gfx1250) — matrix operand order repaired and checked by reading, not compiled, not run on hardware
//
#include <hip/hip_runtime.h>
#include <hip/hip_bf16.h>
#include <math.h>

typedef _Float16 h16_t;
typedef _Float16 v8h  __attribute__((ext_vector_type(8)));
typedef _Float16 v16h __attribute__((ext_vector_type(16)));
typedef float    v8f  __attribute__((ext_vector_type(8)));

#define WMMA_F32_F16(a, b, c) \
    __builtin_amdgcn_wmma_f32_16x16x32_f16(false, (a), false, (b), (short)0, (c), false, false)

// ---------------------------------------------------------------------------
// f32 -> f16 elementwise convert
// ---------------------------------------------------------------------------
__global__ void cvt_f32_to_f16(const float* __restrict__ src, h16_t* __restrict__ dst, long n) {
    long i = (long)blockIdx.x * blockDim.x + threadIdx.x;
    long stride = (long)gridDim.x * blockDim.x;
    for (; i < n; i += stride) dst[i] = (h16_t)src[i];
}

// ---------------------------------------------------------------------------
// f32 [rows,cols] -> f16 [cols,rows] transposed convert (batched)
// ---------------------------------------------------------------------------
__global__ void trans_f32_to_f16(const float* __restrict__ src, h16_t* __restrict__ dst,
                                 int rows, int cols, long sSrc, long sDst) {
    __shared__ float tile[32][33];
    src += (long)blockIdx.z * sSrc;
    dst += (long)blockIdx.z * sDst;
    int c = blockIdx.x * 32 + threadIdx.x;
    for (int i = threadIdx.y; i < 32; i += 8) {
        int r = blockIdx.y * 32 + i;
        if (r < rows && c < cols) tile[i][threadIdx.x] = src[(long)r * cols + c];
    }
    __syncthreads();
    int cc = blockIdx.y * 32 + threadIdx.x;   // output col = original row
    for (int i = threadIdx.y; i < 32; i += 8) {
        int rr = blockIdx.x * 32 + i;         // output row = original col
        if (rr < cols && cc < rows) dst[(long)rr * rows + cc] = (h16_t)tile[threadIdx.x][i];
    }
}

// ---------------------------------------------------------------------------
// Generic WMMA GEMM: C = A[M,K] @ B[K,N] (+bias over N), batched.
// Block = 256 threads = 8 waves (2x4), wave tile 64x64, K step 32.
// Compile-time flags kill all epilogue branching:
//   HAS_BIAS : add bias[col]
//   TRANSC   : C[n*ldc + m] with packed 16B stores (8 contiguous rows per lane)
//   OUT16    : f16 output, else f32
// ---------------------------------------------------------------------------
template <bool HAS_BIAS, bool TRANSC, bool OUT16>
__global__ __launch_bounds__(256) void gemm_f16(
    const h16_t* __restrict__ A, long lda, long sA,
    const h16_t* __restrict__ B, long ldb, long sB,
    h16_t* __restrict__ C16, float* __restrict__ C32, long ldc, long sC,
    const float* __restrict__ bias, int M, int N, int K) {
    int lane = threadIdx.x & 31;
    int w    = threadIdx.x >> 5;
    int wm = w >> 2, wn = w & 3;
    long z = blockIdx.z;
    A += z * sA;
    B += z * sB;
    if (OUT16) C16 += z * sC;
    else       C32 += z * sC;
    long m0 = (long)blockIdx.y * 128 + wm * 64;
    long n0 = (long)blockIdx.x * 256 + wn * 64;
    if (m0 >= M || n0 >= N) return;   // wave-uniform
    int mlo = lane & 15, hi = lane >> 4;

    const h16_t* aRow[4];
#pragma unroll
    for (int mf = 0; mf < 4; mf++) aRow[mf] = A + (m0 + mf * 16 + mlo) * lda;

    v8f acc[4][4] = {};
    for (int k = 0; k < K; k += 32) {
        if (k + 32 < K) {   // hint next K-step into cache (global_prefetch_b8)
            __builtin_prefetch(aRow[0] + k + 32, 0, 0);
            __builtin_prefetch(B + (long)(k + 32 + lane) * ldb + n0, 0, 0);
        }
        v16h a[4], b[4];
#pragma unroll
        for (int mf = 0; mf < 4; mf++) {
            const h16_t* p = aRow[mf] + k + hi * 8;
            ((v8h*)&a[mf])[0] = *(const v8h*)p;
            ((v8h*)&a[mf])[1] = *(const v8h*)(p + 16);
        }
        const h16_t* brow = B + (long)(k + hi * 8) * ldb + n0 + mlo;
#pragma unroll
        for (int nf = 0; nf < 4; nf++) {
            const h16_t* p = brow + nf * 16;
#pragma unroll
            for (int s = 0; s < 16; s++) b[nf][s] = p[(long)(s < 8 ? s : s + 8) * ldb];
        }
#pragma unroll
        for (int mf = 0; mf < 4; mf++)
#pragma unroll
            for (int nf = 0; nf < 4; nf++)
                acc[mf][nf] = WMMA_F32_F16(a[mf], b[nf], acc[mf][nf]);
    }

    // epilogue: branch-free per instantiation
#pragma unroll
    for (int nf = 0; nf < 4; nf++) {
        long col = n0 + nf * 16 + mlo;
        float bb = HAS_BIAS ? bias[col] : 0.0f;
        if (TRANSC) {
            // lane owns 8 contiguous output rows -> one packed 16B store per mf
#pragma unroll
            for (int mf = 0; mf < 4; mf++) {
                v8h pk;
#pragma unroll
                for (int j = 0; j < 8; j++) pk[j] = (h16_t)(acc[mf][nf][j] + bb);
                *(v8h*)(C16 + col * ldc + m0 + mf * 16 + hi * 8) = pk;
            }
        } else {
#pragma unroll
            for (int mf = 0; mf < 4; mf++)
#pragma unroll
                for (int j = 0; j < 8; j++) {
                    long row = m0 + mf * 16 + hi * 8 + j;
                    float val = acc[mf][nf][j] + bb;
                    if (OUT16) C16[row * ldc + col] = (h16_t)val;
                    else       C32[row * ldc + col] = val;
                }
        }
    }
}

// ---------------------------------------------------------------------------
// Causal flash attention, one head per blockIdx.y.
// Each wave: 16 queries x 128 output dims, streams key tiles of 32.
// q  : [2048][2048] f16 (head h at cols h*128..)
// kT : [16][128][2048] f16 (contraction-dim major -> direct B-fragment rows)
// vm : [2048][2048] f16
// ctx: [2048][2048] f16
// ---------------------------------------------------------------------------
__global__ __launch_bounds__(256) void flash_attn(
    const h16_t* __restrict__ q, const h16_t* __restrict__ kT,
    const h16_t* __restrict__ vm, h16_t* __restrict__ ctx) {
    __shared__ h16_t plds[8][16 * 32];
    int lane = threadIdx.x & 31;
    int w    = threadIdx.x >> 5;
    int h    = blockIdx.y;
    int q0   = blockIdx.x * 128 + w * 16;
    int mlo = lane & 15, hi = lane >> 4;
    const float scale = 0.08838834764831845f;   // 1/sqrt(128)

    // persistent Q A-fragments (16 rows x 128 d)
    v16h qa[4];
    const h16_t* qrow = q + (long)(q0 + mlo) * 2048 + h * 128;
#pragma unroll
    for (int kd = 0; kd < 4; kd++) {
        const h16_t* p = qrow + kd * 32 + hi * 8;
        ((v8h*)&qa[kd])[0] = *(const v8h*)p;
        ((v8h*)&qa[kd])[1] = *(const v8h*)(p + 16);
    }

    v8f acc[8] = {};
    float mrun[8], lrun[8];
#pragma unroll
    for (int j = 0; j < 8; j++) { mrun[j] = -1e30f; lrun[j] = 0.0f; }

    const h16_t* kbase = kT + (long)h * 128 * 2048;
    h16_t* myP = &plds[w][0];

    for (int kb = 0; kb <= q0 + 15; kb += 32) {   // wave-uniform causal loop
        // S tile: 16 queries x 32 keys
        v8f s0 = {}, s1 = {};
#pragma unroll
        for (int kd = 0; kd < 4; kd++) {
            const h16_t* krow = kbase + (long)(kd * 32 + hi * 8) * 2048 + kb + mlo;
            v16h b0, b1;
#pragma unroll
            for (int s = 0; s < 16; s++) {
                long o = (long)(s < 8 ? s : s + 8) * 2048;
                b0[s] = krow[o];
                b1[s] = krow[o + 16];
            }
            s0 = WMMA_F32_F16(qa[kd], b0, s0);
            s1 = WMMA_F32_F16(qa[kd], b1, s1);
        }
        // scale + causal mask + online softmax (rows live in 16-lane half-groups)
        float p0[8], p1[8], sc[8];
#pragma unroll
        for (int j = 0; j < 8; j++) {
            int rowg = q0 + hi * 8 + j;
            float a0 = s0[j] * scale, a1 = s1[j] * scale;
            if (kb + mlo      > rowg) a0 = -10000.0f;
            if (kb + 16 + mlo > rowg) a1 = -10000.0f;
            float r = fmaxf(a0, a1);
            for (int m = 1; m < 16; m <<= 1) r = fmaxf(r, __shfl_xor(r, m, 32));
            float nm = fmaxf(mrun[j], r);
            sc[j] = __expf(mrun[j] - nm);
            mrun[j] = nm;
            a0 = __expf(a0 - nm);
            a1 = __expf(a1 - nm);
            float s = a0 + a1;
            for (int m = 1; m < 16; m <<= 1) s += __shfl_xor(s, m, 32);
            lrun[j] = lrun[j] * sc[j] + s;
            p0[j] = a0; p1[j] = a1;
        }
#pragma unroll
        for (int f = 0; f < 8; f++)
#pragma unroll
            for (int j = 0; j < 8; j++) acc[f][j] *= sc[j];

        // C-layout P -> LDS row-major [16 x 32] (private per-wave slab, same-wave
        // LDS ops are in-order; no cross-wave sharing so no barrier needed)
#pragma unroll
        for (int j = 0; j < 8; j++) {
            int row = hi * 8 + j;
            myP[row * 32 + mlo]      = (h16_t)p0[j];
            myP[row * 32 + 16 + mlo] = (h16_t)p1[j];
        }
        // reload as A-fragment (16 x 32)
        v16h pa;
        {
            const h16_t* p = myP + mlo * 32 + hi * 8;
            ((v8h*)&pa)[0] = *(const v8h*)p;
            ((v8h*)&pa)[1] = *(const v8h*)(p + 16);
        }
        // accumulate P @ V
        const h16_t* vrow = vm + (long)(kb + hi * 8) * 2048 + h * 128 + mlo;
#pragma unroll
        for (int f = 0; f < 8; f++) {
            v16h bv;
            const h16_t* p = vrow + f * 16;
#pragma unroll
            for (int s = 0; s < 16; s++) bv[s] = p[(long)(s < 8 ? s : s + 8) * 2048];
            acc[f] = WMMA_F32_F16(pa, bv, acc[f]);
        }
    }
    // epilogue: normalize and store
#pragma unroll
    for (int j = 0; j < 8; j++) {
        float inv = 1.0f / lrun[j];
        long row = q0 + hi * 8 + j;
#pragma unroll
        for (int f = 0; f < 8; f++)
            ctx[row * 2048 + h * 128 + f * 16 + mlo] = (h16_t)(acc[f][j] * inv);
    }
}

// ---------------------------------------------------------------------------
// Host-side orchestration
// ---------------------------------------------------------------------------
static inline void launch_gemm(const h16_t* A, long lda, long sA,
                               const h16_t* B, long ldb, long sB,
                               h16_t* C16, float* C32, long ldc, long sC,
                               const float* bias, int M, int N, int K, int transC,
                               int batch, hipStream_t stream) {
    dim3 g((N + 255) / 256, (M + 127) / 128, batch);
    if (C16) {
        if (transC)
            gemm_f16<false, true, true><<<g, 256, 0, stream>>>(A, lda, sA, B, ldb, sB,
                                                               C16, C32, ldc, sC, bias, M, N, K);
        else if (bias)
            gemm_f16<true, false, true><<<g, 256, 0, stream>>>(A, lda, sA, B, ldb, sB,
                                                               C16, C32, ldc, sC, bias, M, N, K);
        else
            gemm_f16<false, false, true><<<g, 256, 0, stream>>>(A, lda, sA, B, ldb, sB,
                                                                C16, C32, ldc, sC, bias, M, N, K);
    } else {
        gemm_f16<true, false, false><<<g, 256, 0, stream>>>(A, lda, sA, B, ldb, sB,
                                                            C16, C32, ldc, sC, bias, M, N, K);
    }
}

extern "C" void kernel_launch(void* const* d_in, const int* in_sizes, int n_in,
                              void* d_out, int out_size, void* d_ws, size_t ws_size,
                              hipStream_t stream) {
    const float* hid   = (const float*)d_in[0];
    // d_in[1] = attention_mask (causal, known analytically -> unused)
    const float* qkvW  = (const float*)d_in[2];   // [6144, 2048]
    const float* qkvB  = (const float*)d_in[3];   // [6144]
    const float* svdT  = (const float*)d_in[4];   // [2048, 2048]
    const float* svdQK = (const float*)d_in[5];   // [16, 128, 128]
    const float* svdVL = (const float*)d_in[6];   // [16, 128, 128]
    const float* denW  = (const float*)d_in[7];   // [16, 128, 2048]
    const float* denB  = (const float*)d_in[8];   // [2048]
    float* out = (float*)d_out;

    char* ws = (char*)d_ws;
    const size_t MB = 1ull << 20;
    h16_t* h16     = (h16_t*)(ws + 0 * MB);    // 8MB  [0,8)
    h16_t* st16    = (h16_t*)(ws + 8 * MB);    // 8MB  [8,16)
    h16_t* stT16   = (h16_t*)(ws + 16 * MB);   // 8MB  [16,24)
    h16_t* qkvT16  = (h16_t*)(ws + 24 * MB);   // 24MB [24,48)
    h16_t* qk16    = (h16_t*)(ws + 48 * MB);   // 512KB
    h16_t* vlin16  = (h16_t*)(ws + 49 * MB);   // 512KB
    h16_t* vlinT16 = (h16_t*)(ws + 50 * MB);   // 512KB
    h16_t* dw16    = (h16_t*)(ws + 51 * MB);   // 8MB  [51,59)
    h16_t* x16     = (h16_t*)(ws + 59 * MB);   // 8MB  [59,67)
    h16_t* y16     = (h16_t*)(ws + 67 * MB);   // 8MB  [67,75)
    h16_t* mixed16 = (h16_t*)(ws + 75 * MB);   // 24MB [75,99)
    h16_t* qrot16  = h16;      // h16 dead after GEMM1
    h16_t* krotT16 = st16;     // st16 dead after GEMM1
    h16_t* vv16    = stT16;    // stT16 dead after GEMM2
    h16_t* ctx16   = x16;      // x16 dead after GEMM2
    h16_t* tsr16   = y16;      // y16 dead after GEMM3

    // --- converts / transposes ---
    cvt_f32_to_f16<<<2048, 256, 0, stream>>>(hid,   h16,    (long)2048 * 2048);
    cvt_f32_to_f16<<<2048, 256, 0, stream>>>(svdT,  st16,   (long)2048 * 2048);
    cvt_f32_to_f16<<<512,  256, 0, stream>>>(svdQK, qk16,   (long)16 * 128 * 128);
    cvt_f32_to_f16<<<512,  256, 0, stream>>>(svdVL, vlin16, (long)16 * 128 * 128);
    cvt_f32_to_f16<<<2048, 256, 0, stream>>>(denW,  dw16,   (long)16 * 128 * 2048);
    trans_f32_to_f16<<<dim3(64, 64, 1),  dim3(32, 8), 0, stream>>>(svdT, stT16, 2048, 2048, 0, 0);
    trans_f32_to_f16<<<dim3(64, 192, 1), dim3(32, 8), 0, stream>>>(qkvW, qkvT16, 6144, 2048, 0, 0);
    trans_f32_to_f16<<<dim3(4, 4, 16),   dim3(32, 8), 0, stream>>>(svdVL, vlinT16, 128, 128,
                                                                   128 * 128, 128 * 128);

    // --- G1: x = h @ svd_token ---
    launch_gemm(h16, 2048, 0, st16, 2048, 0, x16, nullptr, 2048, 0, nullptr,
                2048, 2048, 2048, 0, 1, stream);
    // --- G2: y = x @ svd_token^T ---
    launch_gemm(x16, 2048, 0, stT16, 2048, 0, y16, nullptr, 2048, 0, nullptr,
                2048, 2048, 2048, 0, 1, stream);
    // --- G3: mixed = y @ qkv_w^T + qkv_b  [2048 x 6144] ---
    launch_gemm(y16, 2048, 0, qkvT16, 6144, 0, mixed16, nullptr, 6144, 0, qkvB,
                2048, 6144, 2048, 0, 1, stream);
    // --- G4: qrot[s, h*128+e] = q_h @ svd_qk[h]  (batched over heads) ---
    launch_gemm(mixed16 + 0,   6144, 384, qk16,   128, 128 * 128,
                qrot16, nullptr, 2048, 128, nullptr, 2048, 128, 128, 0, 16, stream);
    // --- G5: krotT[h][e][s] = (k_h @ svd_qk[h])^T ---
    launch_gemm(mixed16 + 128, 6144, 384, qk16,   128, 128 * 128,
                krotT16, nullptr, 2048, (long)128 * 2048, nullptr, 2048, 128, 128, 1, 16, stream);
    // --- G6: vv[s, h*128+e] = v_h @ svd_vlin[h] ---
    launch_gemm(mixed16 + 256, 6144, 384, vlin16, 128, 128 * 128,
                vv16, nullptr, 2048, 128, nullptr, 2048, 128, 128, 0, 16, stream);
    // --- G7: causal flash attention -> ctx ---
    flash_attn<<<dim3(16, 16), 256, 0, stream>>>(qrot16, krotT16, vv16, ctx16);
    // --- G8: tsr[h*128+e, o] = svd_vlin[h]^T @ dense_w[h] ---
    launch_gemm(vlinT16, 128, 128 * 128, dw16, 2048, (long)128 * 2048,
                tsr16, nullptr, 2048, (long)128 * 2048, nullptr, 128, 2048, 128, 0, 16, stream);
    // --- G9: out = ctx @ tsr + dense_b  (f32 output) ---
    launch_gemm(ctx16, 2048, 0, tsr16, 2048, 0, nullptr, out, 2048, 0, denB,
                2048, 2048, 2048, 0, 1, stream);
}
